// TransformerBlock_28741921145386
// MI455X (gfx1250) — compile-verified, matrix operand order checked by reading, not run on hardware
//
#include <hip/hip_runtime.h>

// ---------------------------------------------------------------------------
// Types for CDNA5 WMMA (wave32, 16x16x32 bf16 -> f32) and TDM descriptors
// ---------------------------------------------------------------------------
typedef __attribute__((ext_vector_type(16))) __bf16 v16bf;
typedef __attribute__((ext_vector_type(8)))  __bf16 v8bf;
typedef __attribute__((ext_vector_type(8)))  float  v8f;
typedef __attribute__((ext_vector_type(4)))  unsigned int u32x4;
typedef __attribute__((ext_vector_type(8)))  int i32x8;
typedef __attribute__((ext_vector_type(4)))  int i32x4;

#define TM 128
#define TN 128
#define TK 32

enum { EPI_F32 = 0, EPI_BF16 = 1, EPI_GELU_BF16 = 2 };

#if __has_builtin(__builtin_amdgcn_tensor_load_to_lds) && \
    __has_builtin(__builtin_amdgcn_s_wait_tensorcnt)
#define USE_TDM 1
#else
#define USE_TDM 0
#endif

// LDS byte offset of a generic pointer known to point into LDS.
static __device__ __forceinline__ unsigned lds_offset(const void* p) {
  return (unsigned)(unsigned long long)(__attribute__((address_space(3))) const void*)p;
}

#if USE_TDM
// Issue one TDM 2-D tile load: tileRows x rowLenElems bf16 elements from a
// row-major tensor (row stride ldElems) into contiguous LDS at ldsAddr.
// OOB rows/cols (beyond tensorDim0/tensorRows) are zero-filled by the TDM.
static __device__ __forceinline__ void tdm_load_2d(
    const __bf16* gsrc, unsigned ldsAddr, unsigned rowLenElems,
    unsigned tileRows, unsigned ldElems, unsigned tensorDim0,
    unsigned tensorRows) {
  unsigned long long ga = (unsigned long long)(size_t)gsrc;
  u32x4 g0;
  g0[0] = 1u;                                         // count=1, user mode
  g0[1] = ldsAddr;                                    // lds_addr
  g0[2] = (unsigned)(ga & 0xffffffffu);               // global_addr[31:0]
  g0[3] = (unsigned)((ga >> 32) & 0x01ffffffu) | 0x80000000u;  // type=2
  unsigned long long s0 = (unsigned long long)ldElems;  // dim0 stride (elems)
  i32x8 g1;
  g1[0] = (int)(1u << 16);                            // data_size=1 (2 bytes)
  g1[1] = (int)((tensorDim0 & 0xffffu) << 16);        // tensor_dim0[15:0]
  g1[2] = (int)((tensorDim0 >> 16) | ((tensorRows & 0xffffu) << 16));
  g1[3] = (int)((tensorRows >> 16) | (rowLenElems << 16));   // tile_dim0
  g1[4] = (int)tileRows;                              // tile_dim1 (tile_dim2=0)
  g1[5] = (int)(unsigned)(s0 & 0xffffffffu);          // dim0_stride[31:0]
  g1[6] = (int)(unsigned)((s0 >> 32) & 0xffffu);      // dim0_stride[47:32]
  g1[7] = 0;
  i32x4 z4 = {0, 0, 0, 0};
  i32x8 z8 = {0, 0, 0, 0, 0, 0, 0, 0};
  // amdgpu-toolchain (clang-23) 6-arg form.
  __builtin_amdgcn_tensor_load_to_lds(g0, g1, z4, z4, z8, 0);
}
#endif

// Load one 16x16x32 A/B fragment from an LDS tile stored row-major [rows][32].
// 16-bit operand layout (ISA 7.12.2): lanes 0-15 hold K={0..7,16..23},
// lanes 16-31 hold K={8..15,24..31}; both halves are 16B-contiguous.
__device__ __forceinline__ v16bf frag16(const __bf16* rowp, int lh) {
  v8bf lo = *reinterpret_cast<const v8bf*>(rowp + lh * 8);
  v8bf hi = *reinterpret_cast<const v8bf*>(rowp + 16 + lh * 8);
  return __builtin_shufflevector(lo, hi, 0, 1, 2, 3, 4, 5, 6, 7,
                                 8, 9, 10, 11, 12, 13, 14, 15);
}

// ---------------------------------------------------------------------------
// Generic batched bf16 GEMM:  C[m,n] = sum_k A[m,k] * Bsrc[n,k]  (+bias[n])
// A: row-major (M x K), lda.  Bsrc: row-major (N x K), ldb (i.e. B^T).
// Batch index z splits as zo = z/innerN, zi = z%innerN with separate strides.
// Staging: TDM DMA (wave 0 issues, TENSORcnt drains) when available,
// otherwise vectorized global->LDS through VGPRs.
// ---------------------------------------------------------------------------
__global__ __launch_bounds__(256) void gemm_bf16(
    const __bf16* __restrict__ A, const __bf16* __restrict__ Bsrc,
    void* __restrict__ Cout, const float* __restrict__ bias,
    int M, int N, int K, int lda, int ldb, int ldc,
    long long sAo, long long sAi, long long sBo, long long sBi,
    long long sCo, long long sCi, int innerN, int epi) {
  int z = blockIdx.z;
  int zo = z / innerN, zi = z % innerN;
  A += (size_t)zo * sAo + (size_t)zi * sAi;
  Bsrc += (size_t)zo * sBo + (size_t)zi * sBi;
  long long cOff = (long long)zo * sCo + (long long)zi * sCi;

  __shared__ __bf16 As[2][TM * TK];
  __shared__ __bf16 Bs[2][TN * TK];

  int tid = threadIdx.x;
  int lane = tid & 31, wave = tid >> 5;
  int waveM = wave >> 2, waveN = wave & 3;   // 2 x 4 wave grid
  int rowBase = blockIdx.y * TM;
  int colBase = blockIdx.x * TN;
  int lr = lane & 15, lh = lane >> 4;

  v8f acc[4][2];
#pragma unroll
  for (int i = 0; i < 4; ++i)
#pragma unroll
    for (int j = 0; j < 2; ++j) {
      v8f zacc = {0.f, 0.f, 0.f, 0.f, 0.f, 0.f, 0.f, 0.f};
      acc[i][j] = zacc;
    }

  int nk = K / TK;

#if USE_TDM
  unsigned asBase = lds_offset(&As[0][0]);
  unsigned bsBase = lds_offset(&Bs[0][0]);
  auto stage = [&](int buf, int k0) {
    if (wave == 0) {
      tdm_load_2d(A + (size_t)rowBase * lda + k0, asBase + buf * (TM * TK * 2),
                  TK, TM, (unsigned)lda, (unsigned)(K - k0),
                  (unsigned)(M - rowBase));
      tdm_load_2d(Bsrc + (size_t)colBase * ldb + k0, bsBase + buf * (TN * TK * 2),
                  TK, TN, (unsigned)ldb, (unsigned)(K - k0),
                  (unsigned)(N - colBase));
    }
  };
  auto stage_wait = [&]() {
    if (wave == 0) __builtin_amdgcn_s_wait_tensorcnt((short)0);
  };
#else
  auto stage = [&](int buf, int k0) {
#pragma unroll
    for (int j = 0; j < 2; ++j) {
      int v = tid + j * 256;          // 512 16-byte vectors per operand tile
      int m = v >> 2, kv = (v & 3) * 8;
      uint4 va = {0u, 0u, 0u, 0u};
      int gr = rowBase + m;
      if (gr < M)
        va = *reinterpret_cast<const uint4*>(A + (size_t)gr * lda + k0 + kv);
      *reinterpret_cast<uint4*>(&As[buf][m * TK + kv]) = va;

      uint4 vb = {0u, 0u, 0u, 0u};
      int gc = colBase + m;           // same index pattern for Bsrc rows (n)
      if (gc < N)
        vb = *reinterpret_cast<const uint4*>(Bsrc + (size_t)gc * ldb + k0 + kv);
      *reinterpret_cast<uint4*>(&Bs[buf][m * TK + kv]) = vb;
    }
  };
  auto stage_wait = [&]() {};
#endif

  stage(0, 0);
  stage_wait();
  __syncthreads();

  for (int ks = 0; ks < nk; ++ks) {
    int cur = ks & 1;
    if (ks + 1 < nk) stage(cur ^ 1, (ks + 1) * TK);
#if !USE_TDM
    if (ks + 2 < nk) {  // speculative prefetch -> global_prefetch_b8
      int m = tid >> 2, kv = (tid & 3) * 8;
      __builtin_prefetch(A + (size_t)(rowBase + m) * lda + (ks + 2) * TK + kv, 0, 3);
      __builtin_prefetch(Bsrc + (size_t)(colBase + m) * ldb + (ks + 2) * TK + kv, 0, 3);
    }
#endif

    const __bf16* At = &As[cur][0];
    const __bf16* Bt = &Bs[cur][0];
    v16bf af[4], bfv[2];
#pragma unroll
    for (int i = 0; i < 4; ++i)
      af[i] = frag16(At + (waveM * 64 + i * 16 + lr) * TK, lh);
#pragma unroll
    for (int j = 0; j < 2; ++j)
      bfv[j] = frag16(Bt + (waveN * 32 + j * 16 + lr) * TK, lh);

#pragma unroll
    for (int i = 0; i < 4; ++i)
#pragma unroll
      for (int j = 0; j < 2; ++j)
        acc[i][j] = __builtin_amdgcn_wmma_f32_16x16x32_bf16(
            false, af[i], false, bfv[j], (short)0, acc[i][j], false, false);

    stage_wait();
    __syncthreads();
  }

  // Epilogue. C/D layout: VGPR e, lanes0-15 -> M=e, lanes16-31 -> M=e+8; N=lane%16.
  float* cf = reinterpret_cast<float*>(Cout);
  __bf16* cb = reinterpret_cast<__bf16*>(Cout);
#pragma unroll
  for (int i = 0; i < 4; ++i)
#pragma unroll
    for (int j = 0; j < 2; ++j) {
      int col = colBase + waveN * 32 + j * 16 + lr;
      if (col >= N) continue;
      float bv = bias ? bias[col] : 0.f;
#pragma unroll
      for (int e = 0; e < 8; ++e) {
        int row = rowBase + waveM * 64 + i * 16 + lh * 8 + e;
        if (row >= M) continue;
        float v = acc[i][j][e] + bv;
        if (epi == EPI_GELU_BF16) v = 0.5f * v * (1.f + erff(v * 0.70710678118f));
        long long idx = cOff + (long long)row * ldc + col;
        if (epi == EPI_F32) cf[idx] = v;
        else                cb[idx] = (__bf16)v;
      }
    }
}

// ---------------------------------------------------------------------------
// LayerNorm stats: deterministic two-stage reduction (sum, sumsq) per batch.
// ---------------------------------------------------------------------------
__global__ __launch_bounds__(256) void reduce_sums(
    const float* __restrict__ x, float* __restrict__ partial, int perBatch) {
  int b = blockIdx.y;
  const float* xb = x + (size_t)b * perBatch;
  float s = 0.f, ss = 0.f;
  for (int i = blockIdx.x * 256 + threadIdx.x; i < perBatch; i += gridDim.x * 256) {
    float v = xb[i];
    s += v;
    ss += v * v;
  }
  for (int o = 16; o; o >>= 1) {
    s += __shfl_xor(s, o, 32);
    ss += __shfl_xor(ss, o, 32);
  }
  __shared__ float rs[8], rss[8];
  int lane = threadIdx.x & 31, w = threadIdx.x >> 5;
  if (lane == 0) { rs[w] = s; rss[w] = ss; }
  __syncthreads();
  if (threadIdx.x == 0) {
    for (int i = 1; i < 8; ++i) { s += rs[i]; ss += rss[i]; }
    partial[(size_t)(b * gridDim.x + blockIdx.x) * 2] = s;
    partial[(size_t)(b * gridDim.x + blockIdx.x) * 2 + 1] = ss;
  }
}

__global__ __launch_bounds__(128) void finalize_stats(
    const float* __restrict__ partial, float* __restrict__ stats, int nParts) {
  __shared__ float rs[128], rss[128];
  int b = blockIdx.x;
  float s = 0.f, ss = 0.f;
  for (int i = threadIdx.x; i < nParts; i += 128) {
    s += partial[(size_t)(b * nParts + i) * 2];
    ss += partial[(size_t)(b * nParts + i) * 2 + 1];
  }
  rs[threadIdx.x] = s; rss[threadIdx.x] = ss;
  __syncthreads();
  for (int o = 64; o; o >>= 1) {
    if (threadIdx.x < o) {
      rs[threadIdx.x] += rs[threadIdx.x + o];
      rss[threadIdx.x] += rss[threadIdx.x + o];
    }
    __syncthreads();
  }
  if (threadIdx.x == 0) { stats[b * 2] = rs[0]; stats[b * 2 + 1] = rss[0]; }
}

// ---------------------------------------------------------------------------
// LN apply over (C,H,W) + transpose (b,c,p) -> row-major (r=b*HW+p, c) bf16.
// Optionally also writes the normalized f32 tensor back in CHW layout.
// ---------------------------------------------------------------------------
__global__ __launch_bounds__(256) void ln_apply_transpose(
    const float* __restrict__ x, const float* __restrict__ stats,
    const float* __restrict__ w, const float* __restrict__ bta,
    __bf16* __restrict__ Abf, float* __restrict__ chw_out,
    int C, int HWn, float invPerBatch) {
  __shared__ float tile[32][33];
  int b = blockIdx.z, c0 = blockIdx.y * 32, p0 = blockIdx.x * 32;
  float s = stats[b * 2], ss = stats[b * 2 + 1];
  float mu = s * invPerBatch;
  float var = fmaxf(ss * invPerBatch - mu * mu, 0.f);
  float rstd = rsqrtf(var + 1e-5f);
  int tx = threadIdx.x & 31, ty = threadIdx.x >> 5;
#pragma unroll
  for (int r = 0; r < 4; ++r) {
    int c = c0 + ty + r * 8, p = p0 + tx;
    size_t gi = ((size_t)b * C + c) * HWn + p;
    size_t wi = (size_t)c * HWn + p;
    float v = (x[gi] - mu) * rstd * w[wi] + bta[wi];
    if (chw_out) chw_out[gi] = v;
    tile[ty + r * 8][tx] = v;  // tile[c_local][p_local]
  }
  __syncthreads();
#pragma unroll
  for (int r = 0; r < 4; ++r) {
    int p = p0 + ty + r * 8, c = c0 + tx;
    Abf[((size_t)b * HWn + p) * C + c] = (__bf16)tile[tx][ty + r * 8];
  }
}

// f32 -> bf16 elementwise (weights).
__global__ __launch_bounds__(256) void f32_to_bf16(
    const float* __restrict__ in, __bf16* __restrict__ out, long long n) {
  for (long long i = (long long)blockIdx.x * 256 + threadIdx.x; i < n;
       i += (long long)gridDim.x * 256)
    out[i] = (__bf16)in[i];
}

// Row softmax over Dq=512 channels (one wave per row), write bf16.
__global__ __launch_bounds__(256) void softmax_rows(
    const float* __restrict__ qkv, int ld, __bf16* __restrict__ out, int Dq) {
  long long r = (long long)blockIdx.x * 8 + (threadIdx.x >> 5);
  int lane = threadIdx.x & 31;
  const float* row = qkv + r * ld;
  float vals[16];
  float m = -3.4e38f;
#pragma unroll
  for (int i = 0; i < 16; ++i) {
    vals[i] = row[lane + i * 32];
    m = fmaxf(m, vals[i]);
  }
  for (int o = 16; o; o >>= 1) m = fmaxf(m, __shfl_xor(m, o, 32));
  float s = 0.f;
#pragma unroll
  for (int i = 0; i < 16; ++i) {
    vals[i] = expf(vals[i] - m);
    s += vals[i];
  }
  for (int o = 16; o; o >>= 1) s += __shfl_xor(s, o, 32);
  float inv = 1.f / s;
#pragma unroll
  for (int i = 0; i < 16; ++i) out[r * Dq + lane + i * 32] = (__bf16)(vals[i] * inv);
}

// Per-(b,c) max and 1/sum(exp) over the spatial dim for K (coalesced reads:
// 32 consecutive channels per row of threads).
__global__ __launch_bounds__(256) void k_colstats(
    const float* __restrict__ qkv, int ld, int colOff,
    float* __restrict__ kmax, float* __restrict__ kdenInv, int C, int HWn) {
  __shared__ float red[8][32];
  int b = blockIdx.y, c0 = blockIdx.x * 32;
  int tx = threadIdx.x & 31, ty = threadIdx.x >> 5;
  const float* base = qkv + (size_t)b * HWn * ld + colOff + c0 + tx;
  float m = -3.4e38f;
  for (int p = ty; p < HWn; p += 8) m = fmaxf(m, base[(size_t)p * ld]);
  red[ty][tx] = m;
  __syncthreads();
  if (ty == 0) {
    for (int i = 1; i < 8; ++i) m = fmaxf(m, red[i][tx]);
    red[0][tx] = m;
  }
  __syncthreads();
  m = red[0][tx];
  __syncthreads();
  float s = 0.f;
  for (int p = ty; p < HWn; p += 8) s += expf(base[(size_t)p * ld] - m);
  red[ty][tx] = s;
  __syncthreads();
  if (ty == 0) {
    for (int i = 1; i < 8; ++i) s += red[i][tx];
    kmax[b * C + c0 + tx] = m;
    kdenInv[b * C + c0 + tx] = 1.f / s;
  }
}

// Transpose a column slice of QKV (r, colOff+c) into (b, c, p) bf16 layout;
// optional exp-normalize (K softmax). Coalesced both directions via LDS.
__global__ __launch_bounds__(256) void qkv_to_bcp(
    const float* __restrict__ qkv, int ld, int colOff,
    const float* __restrict__ kmax, const float* __restrict__ kdenInv,
    __bf16* __restrict__ out, int C, int HWn) {
  __shared__ float tile[32][33];
  int b = blockIdx.z, c0 = blockIdx.y * 32, p0 = blockIdx.x * 32;
  int tx = threadIdx.x & 31, ty = threadIdx.x >> 5;
#pragma unroll
  for (int r = 0; r < 4; ++r) {
    int p = p0 + ty + r * 8, c = c0 + tx;
    float v = qkv[((size_t)b * HWn + p) * ld + colOff + c];
    if (kmax) v = expf(v - kmax[b * C + c]) * kdenInv[b * C + c];
    tile[ty + r * 8][tx] = v;  // tile[p_local][c_local]
  }
  __syncthreads();
#pragma unroll
  for (int r = 0; r < 4; ++r) {
    int c = c0 + ty + r * 8, p = p0 + tx;
    out[((size_t)b * C + c) * HWn + p] = (__bf16)tile[tx][ty + r * 8];
  }
}

// out_chw[b,c,p] = src[r=b*HW+p, c] + res_chw[b,c,p]
__global__ __launch_bounds__(256) void add_res_transpose(
    const float* __restrict__ src, const float* __restrict__ res,
    float* __restrict__ out, int C, int HWn) {
  __shared__ float tile[32][33];
  int b = blockIdx.z, c0 = blockIdx.y * 32, p0 = blockIdx.x * 32;
  int tx = threadIdx.x & 31, ty = threadIdx.x >> 5;
#pragma unroll
  for (int r = 0; r < 4; ++r) {
    int p = p0 + ty + r * 8, c = c0 + tx;
    tile[ty + r * 8][tx] = src[((size_t)b * HWn + p) * C + c];  // tile[p][c]
  }
  __syncthreads();
#pragma unroll
  for (int r = 0; r < 4; ++r) {
    int c = c0 + ty + r * 8, p = p0 + tx;
    size_t gi = ((size_t)b * C + c) * HWn + p;
    out[gi] = tile[tx][ty + r * 8] + res[gi];
  }
}

// ---------------------------------------------------------------------------
// Host orchestration
// ---------------------------------------------------------------------------
extern "C" void kernel_launch(void* const* d_in, const int* in_sizes, int n_in,
                              void* d_out, int out_size, void* d_ws, size_t ws_size,
                              hipStream_t stream) {
  (void)in_sizes; (void)n_in; (void)out_size; (void)ws_size;
  constexpr int kB = 16, kC = 512, kHW = 4096, kDq = 512, kDv = 512, kDh = 2048;
  constexpr int kNqkv = kDq + kDq + kDv;       // 1536
  constexpr long long R = (long long)kB * kHW; // 65536 rows
  constexpr int perBatch = kC * kHW;           // 2,097,152

  const float* x      = (const float*)d_in[0];
  const float* ln1_w  = (const float*)d_in[1];
  const float* ln1_b  = (const float*)d_in[2];
  const float* q_w    = (const float*)d_in[3];
  const float* q_b    = (const float*)d_in[4];
  const float* k_w    = (const float*)d_in[5];
  const float* k_b    = (const float*)d_in[6];
  const float* v_w    = (const float*)d_in[7];
  const float* v_b    = (const float*)d_in[8];
  const float* hu_w   = (const float*)d_in[9];
  const float* hu_b   = (const float*)d_in[10];
  const float* ln2_w  = (const float*)d_in[11];
  const float* ln2_b  = (const float*)d_in[12];
  const float* ff1_w  = (const float*)d_in[13];
  const float* ff1_b  = (const float*)d_in[14];
  const float* ff2_w  = (const float*)d_in[15];
  const float* ff2_b  = (const float*)d_in[16];
  float* out = (float*)d_out;

  char* wsp = (char*)d_ws;
  auto alloc = [&](size_t bytes) {
    char* p = wsp;
    wsp += (bytes + 255) & ~(size_t)255;
    return p;
  };

  float*  partial   = (float*)alloc((size_t)kB * 128 * 2 * 4);
  float*  stats1    = (float*)alloc(kB * 2 * 4);
  float*  stats2    = (float*)alloc(kB * 2 * 4);
  float*  kmax      = (float*)alloc((size_t)kB * kC * 4);
  float*  kdenInv   = (float*)alloc((size_t)kB * kC * 4);
  float*  biasqkv   = (float*)alloc(kNqkv * 4);
  __bf16* Wqkv_bf   = (__bf16*)alloc((size_t)kNqkv * kC * 2);
  __bf16* hu_bf     = (__bf16*)alloc((size_t)kC * kDv * 2);
  __bf16* ff1_bf    = (__bf16*)alloc((size_t)kDh * kC * 2);
  __bf16* ff2_bf    = (__bf16*)alloc((size_t)kC * kDh * 2);
  __bf16* A_bf      = (__bf16*)alloc((size_t)R * kC * 2);
  float*  QKV       = (float*)alloc((size_t)R * kNqkv * 4);
  __bf16* Qs_bf     = (__bf16*)alloc((size_t)R * kDq * 2);
  __bf16* Ks_bf     = (__bf16*)alloc((size_t)R * kDq * 2);   // (b,c,p)
  __bf16* Vt_bf     = (__bf16*)alloc((size_t)R * kDv * 2);   // (b,v,p)
  __bf16* KVt_bf    = (__bf16*)alloc((size_t)kB * 8 * 64 * 64 * 2);
  __bf16* attn_bf   = (__bf16*)alloc((size_t)R * kDv * 2);
  float*  huout     = (float*)alloc((size_t)R * kC * 4);
  float*  x2_chw    = (float*)alloc((size_t)R * kC * 4);
  float*  x2ln_chw  = (float*)alloc((size_t)R * kC * 4);
  __bf16* X2ln_bf   = (__bf16*)alloc((size_t)R * kC * 2);
  __bf16* H_bf      = (__bf16*)alloc((size_t)R * kDh * 2);
  float*  ffout     = (float*)alloc((size_t)R * kC * 4);

  dim3 blk(256);
  dim3 tgrid(kHW / 32, kC / 32, kB);
  float invPB = 1.f / (float)perBatch;

  auto gemm = [&](const __bf16* A, const __bf16* Bm, void* C, const float* bias,
                  int M, int N, int K, int lda, int ldb, int ldc,
                  long long sAo, long long sAi, long long sBo, long long sBi,
                  long long sCo, long long sCi, int innerN, int nbatch, int epi) {
    dim3 g((N + TN - 1) / TN, (M + TM - 1) / TM, nbatch);
    gemm_bf16<<<g, blk, 0, stream>>>(A, Bm, C, bias, M, N, K, lda, ldb, ldc,
                                     sAo, sAi, sBo, sBi, sCo, sCi, innerN, epi);
  };

  // ---- LN1 ----
  reduce_sums<<<dim3(128, kB), blk, 0, stream>>>(x, partial, perBatch);
  finalize_stats<<<dim3(kB), dim3(128), 0, stream>>>(partial, stats1, 128);
  ln_apply_transpose<<<tgrid, blk, 0, stream>>>(x, stats1, ln1_w, ln1_b,
                                                A_bf, nullptr, kC, kHW, invPB);

  // ---- weights -> bf16 (already in the (N rows x K cols) layout we need) ----
  f32_to_bf16<<<dim3(512), blk, 0, stream>>>(q_w, Wqkv_bf, (long long)kDq * kC);
  f32_to_bf16<<<dim3(512), blk, 0, stream>>>(k_w, Wqkv_bf + (size_t)kDq * kC, (long long)kDq * kC);
  f32_to_bf16<<<dim3(512), blk, 0, stream>>>(v_w, Wqkv_bf + (size_t)2 * kDq * kC, (long long)kDv * kC);
  f32_to_bf16<<<dim3(512), blk, 0, stream>>>(hu_w, hu_bf, (long long)kC * kDv);
  f32_to_bf16<<<dim3(512), blk, 0, stream>>>(ff1_w, ff1_bf, (long long)kDh * kC);
  f32_to_bf16<<<dim3(512), blk, 0, stream>>>(ff2_w, ff2_bf, (long long)kC * kDh);
  (void)hipMemcpyAsync(biasqkv,           q_b, kDq * 4, hipMemcpyDeviceToDevice, stream);
  (void)hipMemcpyAsync(biasqkv + kDq,     k_b, kDq * 4, hipMemcpyDeviceToDevice, stream);
  (void)hipMemcpyAsync(biasqkv + 2 * kDq, v_b, kDv * 4, hipMemcpyDeviceToDevice, stream);

  // ---- fused QKV projection (65536x1536x512), f32 out ----
  gemm(A_bf, Wqkv_bf, QKV, biasqkv, (int)R, kNqkv, kC, kC, kC, kNqkv,
       0, 0, 0, 0, 0, 0, 1, 1, EPI_F32);

  // ---- softmaxes ----
  softmax_rows<<<dim3((unsigned)(R / 8)), blk, 0, stream>>>(QKV, kNqkv, Qs_bf, kDq);
  k_colstats<<<dim3(kC / 32, kB), blk, 0, stream>>>(QKV, kNqkv, kDq, kmax, kdenInv, kC, kHW);
  qkv_to_bcp<<<tgrid, blk, 0, stream>>>(QKV, kNqkv, kDq, kmax, kdenInv, Ks_bf, kC, kHW);
  qkv_to_bcp<<<tgrid, blk, 0, stream>>>(QKV, kNqkv, 2 * kDq, nullptr, nullptr, Vt_bf, kC, kHW);

  // ---- KV^T[v,k] = sum_p V[v,p] K[k,p] : batched over 128 (b,h), K=4096 ----
  gemm(Vt_bf, Ks_bf, KVt_bf, nullptr, 64, 64, kHW, kHW, kHW, 64,
       64LL * kHW, 0, 64LL * kHW, 0, 64LL * 64, 0, 1, kB * 8, EPI_BF16);

  // ---- attn[r,v] = sum_k Qs[r,k] KV^T[v,k] : batched (b,h), M=4096, N=K=64 ----
  gemm(Qs_bf, KVt_bf, attn_bf, nullptr, kHW, 64, 64, kDq, 64, kDv,
       (long long)kHW * kDq, 64, 8LL * 64 * 64, 64LL * 64,
       (long long)kHW * kDv, 64, 8, kB * 8, EPI_BF16);

  // ---- hu projection, then residual + transpose back to CHW ----
  gemm(attn_bf, hu_bf, huout, hu_b, (int)R, kC, kDv, kDv, kDv, kC,
       0, 0, 0, 0, 0, 0, 1, 1, EPI_F32);
  add_res_transpose<<<tgrid, blk, 0, stream>>>(huout, x, x2_chw, kC, kHW);

  // ---- LN2 (keeps CHW f32 copy for final residual + bf16 GEMM operand) ----
  reduce_sums<<<dim3(128, kB), blk, 0, stream>>>(x2_chw, partial, perBatch);
  finalize_stats<<<dim3(kB), dim3(128), 0, stream>>>(partial, stats2, 128);
  ln_apply_transpose<<<tgrid, blk, 0, stream>>>(x2_chw, stats2, ln2_w, ln2_b,
                                                X2ln_bf, x2ln_chw, kC, kHW, invPB);

  // ---- FFN ----
  gemm(X2ln_bf, ff1_bf, H_bf, ff1_b, (int)R, kDh, kC, kC, kC, kDh,
       0, 0, 0, 0, 0, 0, 1, 1, EPI_GELU_BF16);
  gemm(H_bf, ff2_bf, ffout, ff2_b, (int)R, kC, kDh, kDh, kDh, kC,
       0, 0, 0, 0, 0, 0, 1, 1, EPI_F32);

  // ---- final residual (post-LN2 x) + transpose to (B,C,H,W) output ----
  add_res_transpose<<<tgrid, blk, 0, stream>>>(ffout, x2ln_chw, out, kC, kHW);
}
